// SSAN_SAM_DWAP_3874060501768
// MI455X (gfx1250) — matrix operand order repaired and checked by reading, not compiled, not run on hardware
//
#include <hip/hip_runtime.h>
#include <hip/hip_bf16.h>

// ---------------- model constants ----------------
#define B_   8
#define C_   684
#define H_   16
#define W_   64
#define HW_  1024
#define N_   8192        // B*H*W positions
#define T_   32
#define HID_ 256
#define INP_ 256
#define ATT_ 512
#define COV_ 512
#define VOC_ 111
#define LOC_ 432
#define KCOV_ 11
#define KK_   121        // 11*11
#define KPAD_ 128        // padded im2col K
#define CPAD_ 704        // 684 padded to 22*32

typedef __attribute__((ext_vector_type(16))) _Float16 v16h;
typedef __attribute__((ext_vector_type(8)))  _Float16 v8h;
typedef __attribute__((ext_vector_type(8)))  float    v8f;

// ---------------- helpers ----------------
__device__ inline float block_reduce_sum(float v, float* sbuf) {
    int tid = threadIdx.x;
    sbuf[tid] = v; __syncthreads();
    for (int s = blockDim.x >> 1; s > 0; s >>= 1) {
        if (tid < s) sbuf[tid] += sbuf[tid + s];
        __syncthreads();
    }
    float r = sbuf[0]; __syncthreads();
    return r;
}

__device__ inline float sigmoidf_(float x) { return 1.f / (1.f + expf(-x)); }

// ---------------- precompute kernels ----------------

// mask_hw[b,hw] = images_mask[b,0,h*16,w*16]; masksum[b] = sum
__global__ void k_mask(const float* __restrict__ im, float* __restrict__ mask_hw,
                       float* __restrict__ masksum) {
    __shared__ float sbuf[256];
    int b = blockIdx.x, tid = threadIdx.x;
    float s = 0.f;
    for (int j = tid; j < HW_; j += 256) {
        int h = j >> 6, w = j & 63;
        float v = im[(size_t)b * 256 * 1024 + (size_t)(h * 16) * 1024 + (size_t)w * 16];
        mask_hw[b * HW_ + j] = v;
        s += v;
    }
    s = block_reduce_sum(s, sbuf);
    if (tid == 0) masksum[b] = s;
}

// avg[b,c] = sum_hw feat*mask / masksum
__global__ void k_avg(const float* __restrict__ feat, const float* __restrict__ mask_hw,
                      const float* __restrict__ masksum, float* __restrict__ avg) {
    __shared__ float sbuf[256];
    int c = blockIdx.x, b = blockIdx.y, tid = threadIdx.x;
    const float* f = feat + ((size_t)b * C_ + c) * HW_;
    const float* mk = mask_hw + b * HW_;
    float s = 0.f;
    for (int j = tid; j < HW_; j += 256) s += f[j] * mk[j];
    s = block_reduce_sum(s, sbuf);
    if (tid == 0) avg[b * C_ + c] = s / masksum[b];
}

// generic small linear: out[b,o] = act( x[b,:] . W[o,:] + bias[o] )
__global__ void k_linear(const float* __restrict__ x, const float* __restrict__ W,
                         const float* __restrict__ bias, float* __restrict__ out,
                         int Bn, int K, int O, int act) {
    int idx = blockIdx.x * blockDim.x + threadIdx.x;
    if (idx >= Bn * O) return;
    int b = idx / O, o = idx % O;
    float acc = bias ? bias[o] : 0.f;
    const float* xr = x + (size_t)b * K;
    const float* wr = W + (size_t)o * K;
    for (int i = 0; i < K; ++i) acc += xr[i] * wr[i];
    if (act == 1) acc = tanhf(acc);
    out[idx] = acc;
}

// K_fused[a,k] = sum_c W_att[a,c] * K_cov[c, k]   (f16, padded to 128)
__global__ void k_kfused(const float* __restrict__ W_att, const float* __restrict__ K_cov,
                         _Float16* __restrict__ Kf) {
    int idx = blockIdx.x * 256 + threadIdx.x;
    if (idx >= ATT_ * KPAD_) return;
    int a = idx >> 7, k = idx & 127;
    float s = 0.f;
    if (k < KK_)
        for (int c = 0; c < COV_; ++c) s += W_att[a * COV_ + c] * K_cov[c * KK_ + k];
    Kf[idx] = (_Float16)s;
}

// feat_nc_h[n, c] (f16, c padded to 704) = cnn_features[b,c,hw]
__global__ void k_feat_nc(const float* __restrict__ feat, _Float16* __restrict__ out) {
    size_t idx = (size_t)blockIdx.x * 256 + threadIdx.x;
    if (idx >= (size_t)N_ * CPAD_) return;
    int n = (int)(idx / CPAD_), c = (int)(idx % CPAD_);
    int b = n >> 10, hw = n & 1023;
    float v = (c < C_) ? feat[((size_t)b * C_ + c) * HW_ + hw] : 0.f;
    out[idx] = (_Float16)v;
}

// KencH[a, c] (f16, padded) = K_enc[a, c]
__global__ void k_kenc(const float* __restrict__ K_enc, _Float16* __restrict__ out) {
    int idx = blockIdx.x * 256 + threadIdx.x;
    if (idx >= ATT_ * CPAD_) return;
    int a = idx / CPAD_, c = idx % CPAD_;
    float v = (c < C_) ? K_enc[a * C_ + c] : 0.f;
    out[idx] = (_Float16)v;
}

__global__ void k_zero(float* __restrict__ p, int n) {
    int i = blockIdx.x * 256 + threadIdx.x;
    if (i < n) p[i] = 0.f;
}

// One-time encoder GEMM: trans_p[n,a] = sum_c featH[n,c]*KencH[a,c] + b_enc[a]
// M = 8192 (n), N = 512 (a), K = 704.  One wave per 16x16 output tile.
__global__ __launch_bounds__(32) void k_gemm_trans(const _Float16* __restrict__ A,
                                                   const _Float16* __restrict__ Bm,
                                                   const float* __restrict__ bias,
                                                   float* __restrict__ out) {
    const int nt = blockIdx.x;   // 0..511
    const int at = blockIdx.y;   // 0..31
    const int lane = threadIdx.x;
    const int m = lane & 15, g = lane >> 4;
    const int n = nt * 16 + m;
    const int a = at * 16 + m;
    v8f acc = {};
    for (int c = 0; c < CPAD_ / 32; ++c) {
        // A fragment (16x32 f16): lane m, halves 0..15 -> k = c*32 + 16g + h  (contiguous)
        const _Float16* ap = A + (size_t)n * CPAD_ + c * 32 + 16 * g;
        v8h lo = *(const v8h*)ap;
        v8h hi = *(const v8h*)(ap + 8);
        v16h av;
        #pragma unroll
        for (int i = 0; i < 8; ++i) { av[i] = lo[i]; av[8 + i] = hi[i]; }
        // B fragment (32x16 f16): lane col = m, halves k = c*32 + 16g + h  (contiguous)
        v16h bv = *(const v16h*)(Bm + (size_t)a * CPAD_ + c * 32 + 16 * g);
        acc = __builtin_amdgcn_wmma_f32_16x16x32_f16(false, av, false, bv,
                                                     (short)0, acc, false, false);
    }
    const float bb = bias[a];
    #pragma unroll
    for (int i = 0; i < 8; ++i) {
        int row = nt * 16 + i + 8 * g;
        out[(size_t)row * ATT_ + a] = acc[i] + bb;
    }
}

// ---------------- per-step kernels ----------------

// GRU cell: hidden_new from hidden_prev and embedded token
__global__ void k_gru(const float* __restrict__ hp, float* __restrict__ hn,
                      const int* __restrict__ labels, const float* __restrict__ emb,
                      const float* __restrict__ W_ih, const float* __restrict__ b_ih,
                      const float* __restrict__ W_hh, const float* __restrict__ b_hh,
                      int t) {
    int b = blockIdx.x, o = threadIdx.x;   // block 256
    int tok = (t == 0) ? 1 : labels[b * T_ + (t - 1)];
    const float* we = emb + (size_t)tok * INP_;
    const float* h = hp + b * HID_;
    float gir = b_ih[o], giz = b_ih[HID_ + o], gin = b_ih[2 * HID_ + o];
    float ghr = b_hh[o], ghz = b_hh[HID_ + o], ghn = b_hh[2 * HID_ + o];
    for (int i = 0; i < INP_; ++i) {
        float w = we[i];
        gir += w * W_ih[(size_t)o * INP_ + i];
        giz += w * W_ih[(size_t)(HID_ + o) * INP_ + i];
        gin += w * W_ih[(size_t)(2 * HID_ + o) * INP_ + i];
    }
    for (int i = 0; i < HID_; ++i) {
        float hv = h[i];
        ghr += hv * W_hh[(size_t)o * HID_ + i];
        ghz += hv * W_hh[(size_t)(HID_ + o) * HID_ + i];
        ghn += hv * W_hh[(size_t)(2 * HID_ + o) * HID_ + i];
    }
    float r = sigmoidf_(gir + ghr);
    float z = sigmoidf_(giz + ghz);
    float nn = tanhf(gin + r * ghn);
    hn[b * HID_ + o] = (1.f - z) * nn + z * h[o];
}

// Fused coverage-conv (as WMMA im2col GEMM with pre-fused kernel) + tanh-attention energy.
// energy[n] = b_ac + sum_a W_ac[a] * tanh( (patch(n).Kf[a]) + query[b,a] + trans_p[n,a] )
// 4 waves/block; each wave owns 8 of the 32 a-tiles; LDS im2col patch shared.
__global__ __launch_bounds__(128) void k_attention(const float* __restrict__ alpha_sum,
                                                   const _Float16* __restrict__ Kf,
                                                   const float* __restrict__ query,
                                                   const float* __restrict__ trans_p,
                                                   const float* __restrict__ W_ac,
                                                   const float* __restrict__ b_ac,
                                                   float* __restrict__ energy) {
    __shared__ float patch[11 * 26];
    __shared__ float eSum[16];
    const int tile = blockIdx.x;            // 0..511: 16 consecutive w positions
    const int b  = tile >> 6;
    const int h  = (tile & 63) >> 2;
    const int w0 = (tile & 3) << 4;
    const int tid = threadIdx.x;
    const int lane = tid & 31;
    const int wave = tid >> 5;

    // stage zero-padded 11x26 alpha_sum patch in LDS
    for (int i = tid; i < 11 * 26; i += 128) {
        int r = i / 26, cc = i % 26;
        int ar = h - 5 + r, ac = w0 - 5 + cc;
        float v = 0.f;
        if (ar >= 0 && ar < H_ && ac >= 0 && ac < W_) v = alpha_sum[b * HW_ + ar * W_ + ac];
        patch[i] = v;
    }
    if (tid < 16) eSum[tid] = b_ac[0];
    __syncthreads();

    const int m = lane & 15;
    const int g = lane >> 4;

    // Build A fragments (im2col rows, K padded 121->128) in contiguous k order (k = c*32 + 16g + hh).
    v16h afr[4];
    #pragma unroll
    for (int c = 0; c < 4; ++c) {
        #pragma unroll
        for (int hh = 0; hh < 16; ++hh) {
            int k = c * 32 + hh + 16 * g;
            float v = 0.f;
            if (k < KK_) {
                int dy = k / KCOV_, dx = k % KCOV_;
                v = patch[dy * 26 + m + dx];
            }
            afr[c][hh] = (_Float16)v;
        }
    }

    const int n0 = b * HW_ + h * W_ + w0;
    float accE[8];
    #pragma unroll
    for (int i = 0; i < 8; ++i) accE[i] = 0.f;

    for (int at = wave; at < ATT_ / 16; at += 4) {
        const int a = at * 16 + m;          // this lane's att column
        v8f acc = {};
        #pragma unroll
        for (int c = 0; c < 4; ++c) {
            v16h bfr = *(const v16h*)(Kf + (size_t)a * KPAD_ + c * 32 + 16 * g);
            acc = __builtin_amdgcn_wmma_f32_16x16x32_f16(false, afr[c], false, bfr,
                                                         (short)0, acc, false, false);
        }
        const float qv  = query[b * ATT_ + a];
        const float wac = W_ac[a];
        #pragma unroll
        for (int i = 0; i < 8; ++i) {
            int row = i + 8 * g;            // position offset within tile
            float s = acc[i] + qv + trans_p[(size_t)(n0 + row) * ATT_ + a];
            accE[i] += wac * tanhf(s);
        }
    }

    // reduce over the 16 att columns held by this half-wave, then over waves
    #pragma unroll
    for (int i = 0; i < 8; ++i) {
        float v = accE[i];
        v += __shfl_xor(v, 1);
        v += __shfl_xor(v, 2);
        v += __shfl_xor(v, 4);
        v += __shfl_xor(v, 8);
        if (m == 0) atomicAdd(&eSum[i + 8 * g], v);
    }
    __syncthreads();
    if (tid < 16) energy[n0 + tid] = eSum[tid];
}

// softmax over hw per batch (per-b max-shift == reference's global shift after normalize),
// alpha>0.02 gating, alpha_sum accumulation, and word_alphas output write.
__global__ void k_softmax(const float* __restrict__ energy, const float* __restrict__ mask_hw,
                          float* __restrict__ alpha_sum, float* __restrict__ alpha_eff,
                          float* __restrict__ out_alpha, int t) {
    __shared__ float sbuf[256];
    int b = blockIdx.x, tid = threadIdx.x;
    float e[4]; float mx = -1e30f;
    #pragma unroll
    for (int j = 0; j < 4; ++j) {
        e[j] = energy[b * HW_ + tid + j * 256];
        mx = fmaxf(mx, e[j]);
    }
    sbuf[tid] = mx; __syncthreads();
    for (int s = 128; s > 0; s >>= 1) { if (tid < s) sbuf[tid] = fmaxf(sbuf[tid], sbuf[tid + s]); __syncthreads(); }
    mx = sbuf[0]; __syncthreads();
    float ev[4]; float sum = 0.f;
    #pragma unroll
    for (int j = 0; j < 4; ++j) {
        int hw = tid + j * 256;
        ev[j] = expf(e[j] - mx) * mask_hw[b * HW_ + hw];
        sum += ev[j];
    }
    sum = block_reduce_sum(sum, sbuf) + 1e-10f;
    #pragma unroll
    for (int j = 0; j < 4; ++j) {
        int hw = tid + j * 256;
        float al = ev[j] / sum;
        alpha_eff[b * HW_ + hw] = (al > 0.02f) ? al : 0.f;
        alpha_sum[b * HW_ + hw] += al;
        out_alpha[((size_t)b * T_ + t) * HW_ + hw] = al;
    }
}

// ctx[b,c] = sum_hw alpha_eff[b,hw] * feat[b,c,hw]
__global__ void k_ctx(const float* __restrict__ alpha_eff, const float* __restrict__ feat,
                      float* __restrict__ ctx) {
    __shared__ float sbuf[256];
    int c = blockIdx.x, b = blockIdx.y, tid = threadIdx.x;
    const float* f = feat + ((size_t)b * C_ + c) * HW_;
    const float* ae = alpha_eff + b * HW_;
    float s = 0.f;
    for (int j = tid; j < HW_; j += 256) s += ae[j] * f[j];
    s = block_reduce_sum(s, sbuf);
    if (tid == 0) ctx[b * C_ + c] = s;
}

// out_state = max(hid.W_state^T + b_state + we.W_embw^T + b_embw + ctx.W_ctx^T + b_ctx, loc_w)
__global__ void k_outstate(const float* __restrict__ hid, const int* __restrict__ labels,
                           const float* __restrict__ emb, const float* __restrict__ ctx,
                           const float* __restrict__ loc_w,
                           const float* __restrict__ W_state, const float* __restrict__ b_state,
                           const float* __restrict__ W_embw,  const float* __restrict__ b_embw,
                           const float* __restrict__ W_ctx,   const float* __restrict__ b_ctx,
                           float* __restrict__ out_state, int t) {
    int b = blockIdx.x, o = threadIdx.x;
    int tok = (t == 0) ? 1 : labels[b * T_ + (t - 1)];
    const float* we = emb + (size_t)tok * INP_;
    const float* h = hid + b * HID_;
    float acc = b_state[o] + b_embw[o] + b_ctx[o];
    for (int i = 0; i < HID_; ++i)
        acc += h[i] * W_state[(size_t)o * HID_ + i] + we[i] * W_embw[(size_t)o * INP_ + i];
    const float* cx = ctx + b * C_;
    for (int c = 0; c < C_; ++c) acc += cx[c] * W_ctx[(size_t)o * C_ + c];
    out_state[b * HID_ + o] = fmaxf(acc, loc_w[b * HID_ + o]);
}

__global__ void k_prob(const float* __restrict__ out_state, const float* __restrict__ W_out,
                       const float* __restrict__ b_out, float* __restrict__ word_probs, int t) {
    int b = blockIdx.x, v = threadIdx.x;  // block 128
    if (v >= VOC_) return;
    const float* x = out_state + b * HID_;
    float s = b_out[v];
    for (int i = 0; i < HID_; ++i) s += x[i] * W_out[(size_t)v * HID_ + i];
    word_probs[((size_t)b * T_ + t) * VOC_ + v] = s;
}

// ---------------- launcher ----------------
extern "C" void kernel_launch(void* const* d_in, const int* in_sizes, int n_in,
                              void* d_out, int out_size, void* d_ws, size_t ws_size,
                              hipStream_t stream) {
    const float* feat     = (const float*)d_in[0];
    const int*   labels   = (const int*)  d_in[1];
    const float* loc_pred = (const float*)d_in[2];
    const float* im_mask  = (const float*)d_in[3];
    // d_in[4] labels_mask unused (unused in reference)
    const float* W_init = (const float*)d_in[5];
    const float* b_init = (const float*)d_in[6];
    const float* emb    = (const float*)d_in[7];
    const float* W_ih   = (const float*)d_in[8];
    const float* b_ih   = (const float*)d_in[9];
    const float* W_hh   = (const float*)d_in[10];
    const float* b_hh   = (const float*)d_in[11];
    const float* W_hq   = (const float*)d_in[12];
    const float* b_hq   = (const float*)d_in[13];
    const float* K_cov  = (const float*)d_in[14];
    const float* W_att  = (const float*)d_in[15];
    const float* W_ac   = (const float*)d_in[16];
    const float* b_ac   = (const float*)d_in[17];
    const float* K_enc  = (const float*)d_in[18];
    const float* b_enc  = (const float*)d_in[19];
    const float* W_state = (const float*)d_in[20];
    const float* b_state = (const float*)d_in[21];
    const float* W_embw  = (const float*)d_in[22];
    const float* b_embw  = (const float*)d_in[23];
    const float* W_ctx   = (const float*)d_in[24];
    const float* b_ctx   = (const float*)d_in[25];
    const float* W_out   = (const float*)d_in[26];
    const float* b_out   = (const float*)d_in[27];
    const float* W_loc   = (const float*)d_in[28];
    const float* b_loc   = (const float*)d_in[29];

    float* word_probs  = (float*)d_out;                       // (B,T,VOC)
    float* word_alphas = word_probs + (size_t)B_ * T_ * VOC_; // (B,T,H,W)

    // workspace carve (aligned 256B)
    char* p = (char*)d_ws;
    auto alloc = [&](size_t bytes) { void* r = p; p += (bytes + 255) & ~(size_t)255; return r; };
    float* mask_hw   = (float*)alloc((size_t)B_ * HW_ * 4);
    float* masksum   = (float*)alloc(B_ * 4);
    float* avg       = (float*)alloc((size_t)B_ * C_ * 4);
    float* hbuf0     = (float*)alloc((size_t)B_ * HID_ * 4);
    float* hbuf1     = (float*)alloc((size_t)B_ * HID_ * 4);
    float* loc_w     = (float*)alloc((size_t)B_ * HID_ * 4);
    float* query     = (float*)alloc((size_t)B_ * ATT_ * 4);
    float* energy    = (float*)alloc((size_t)N_ * 4);
    float* alpha_sum = (float*)alloc((size_t)N_ * 4);
    float* alpha_eff = (float*)alloc((size_t)N_ * 4);
    float* ctx       = (float*)alloc((size_t)B_ * C_ * 4);
    float* out_state = (float*)alloc((size_t)B_ * HID_ * 4);
    _Float16* Kf     = (_Float16*)alloc((size_t)ATT_ * KPAD_ * 2);
    _Float16* KencH  = (_Float16*)alloc((size_t)ATT_ * CPAD_ * 2);
    _Float16* featH  = (_Float16*)alloc((size_t)N_ * CPAD_ * 2);
    float* trans_p   = (float*)alloc((size_t)N_ * ATT_ * 4);

    // ---- one-time precompute ----
    k_mask<<<B_, 256, 0, stream>>>(im_mask, mask_hw, masksum);
    k_avg<<<dim3(C_, B_), 256, 0, stream>>>(feat, mask_hw, masksum, avg);
    k_linear<<<(B_ * HID_ + 255) / 256, 256, 0, stream>>>(avg, W_init, b_init, hbuf0, B_, C_, HID_, 1);
    k_linear<<<(B_ * HID_ + 255) / 256, 256, 0, stream>>>(loc_pred, W_loc, b_loc, loc_w, B_, LOC_, HID_, 0);
    k_kfused<<<(ATT_ * KPAD_ + 255) / 256, 256, 0, stream>>>(W_att, K_cov, Kf);
    k_feat_nc<<<(int)(((size_t)N_ * CPAD_ + 255) / 256), 256, 0, stream>>>(feat, featH);
    k_kenc<<<(ATT_ * CPAD_ + 255) / 256, 256, 0, stream>>>(K_enc, KencH);
    k_gemm_trans<<<dim3(N_ / 16, ATT_ / 16), 32, 0, stream>>>(featH, KencH, b_enc, trans_p);
    k_zero<<<(N_ + 255) / 256, 256, 0, stream>>>(alpha_sum, N_);

    // ---- decode loop ----
    float* hbuf[2] = {hbuf0, hbuf1};
    for (int t = 0; t < T_; ++t) {
        const float* hp = hbuf[t & 1];
        float*       hn = hbuf[(t + 1) & 1];
        k_gru<<<B_, HID_, 0, stream>>>(hp, hn, labels, emb, W_ih, b_ih, W_hh, b_hh, t);
        k_linear<<<(B_ * ATT_ + 255) / 256, 256, 0, stream>>>(hn, W_hq, b_hq, query, B_, HID_, ATT_, 0);
        k_attention<<<N_ / 16, 128, 0, stream>>>(alpha_sum, Kf, query, trans_p, W_ac, b_ac, energy);
        k_softmax<<<B_, 256, 0, stream>>>(energy, mask_hw, alpha_sum, alpha_eff, word_alphas, t);
        k_ctx<<<dim3(C_, B_), 256, 0, stream>>>(alpha_eff, feat, ctx);
        k_outstate<<<B_, HID_, 0, stream>>>(hn, labels, emb, ctx, loc_w,
                                            W_state, b_state, W_embw, b_embw, W_ctx, b_ctx,
                                            out_state, t);
        k_prob<<<B_, 128, 0, stream>>>(out_state, W_out, b_out, word_probs, t);
    }
}
